// EncoderBlock_13125420056849
// MI455X (gfx1250) — compile-verified, matrix operand order checked by reading, not run on hardware
//
#include <hip/hip_runtime.h>
#include <hip/hip_bf16.h>
#include <math.h>

// ---------------------------------------------------------------------------
// QANet-style encoder block for MI455X (gfx1250, wave32, WMMA).
// All matmuls via v_wmma_f32_16x16x32_f16; fragments loaded as b128.
// ---------------------------------------------------------------------------

typedef __attribute__((ext_vector_type(16))) _Float16 v16h;
typedef __attribute__((ext_vector_type(8)))  _Float16 v8h;
typedef __attribute__((ext_vector_type(4)))  _Float16 v4h;
typedef __attribute__((ext_vector_type(8)))  float    v8f;

union H16 { v16h f; v8h h[2]; };

#define Bdim 16
#define Sdim 512
#define Ddim 128
#define Hdim 8
#define NCdim 4
#define Kdim 7
#define TOKS (Bdim * Sdim)          // 8192 token rows
#define TOKD (TOKS * Ddim)          // 1,048,576 elements

// ---------------------------------------------------------------------------
// Positional signal add: cur = x + pos(S, D)   (float4 vectorized)
// ---------------------------------------------------------------------------
__global__ void pos_add_kernel(const float* __restrict__ x, float* __restrict__ out) {
    int idx = blockIdx.x * blockDim.x + threadIdx.x;       // over TOKD/4
    if (idx >= TOKD / 4) return;
    int c4 = (idx & 31) * 4;
    int s  = (idx >> 5) & (Sdim - 1);
    const int nt = Ddim / 2;
    float inc = __logf(10000.0f) / (float)(nt - 1);
    float4 xv = ((const float4*)x)[idx];
    float pr[4];
#pragma unroll
    for (int i = 0; i < 4; ++i) {
        int c = c4 + i;
        pr[i] = (c < nt) ? __sinf((float)s * __expf(-(float)c * inc))
                         : __cosf((float)s * __expf(-(float)(c - nt) * inc));
    }
    float4 o = { xv.x + pr[0], xv.y + pr[1], xv.z + pr[2], xv.w + pr[3] };
    ((float4*)out)[idx] = o;
}

// ---------------------------------------------------------------------------
// LayerNorm over D=128: one wave32 per row, 4 channels/lane, shfl reductions.
// ---------------------------------------------------------------------------
__global__ __launch_bounds__(32)
void ln_kernel(const float* __restrict__ x, const float* __restrict__ g,
               const float* __restrict__ b, float* __restrict__ outf,
               _Float16* __restrict__ outh) {
    size_t row = blockIdx.x;
    int lane = threadIdx.x;
    float4 v = ((const float4*)(x + row * Ddim))[lane];
    float s = v.x + v.y + v.z + v.w;
#pragma unroll
    for (int o = 16; o > 0; o >>= 1) s += __shfl_xor(s, o, 32);
    float mu = s * (1.0f / 128.0f);
    float4 d = { v.x - mu, v.y - mu, v.z - mu, v.w - mu };
    float q = d.x * d.x + d.y * d.y + d.z * d.z + d.w * d.w;
#pragma unroll
    for (int o = 16; o > 0; o >>= 1) q += __shfl_xor(q, o, 32);
    float rs = rsqrtf(q * (1.0f / 128.0f) + 1e-5f);
    float4 gv = ((const float4*)g)[lane];
    float4 bv = ((const float4*)b)[lane];
    float y0 = d.x * rs * gv.x + bv.x;
    float y1 = d.y * rs * gv.y + bv.y;
    float y2 = d.z * rs * gv.z + bv.z;
    float y3 = d.w * rs * gv.w + bv.w;
    if (outf) {
        float4 o4 = { y0, y1, y2, y3 };
        ((float4*)(outf + row * Ddim))[lane] = o4;
    }
    if (outh) {
        v4h h4 = { (_Float16)y0, (_Float16)y1, (_Float16)y2, (_Float16)y3 };
        *(v4h*)(outh + row * Ddim + lane * 4) = h4;
    }
}

// ---------------------------------------------------------------------------
// Depthwise conv along S (K=7, zero pad), + bias, 4 channels per thread.
// Output f16 (GEMM A operand).
// ---------------------------------------------------------------------------
__global__ void dwconv_kernel(const float* __restrict__ h, const float* __restrict__ w,
                              const float* __restrict__ bias, _Float16* __restrict__ out) {
    int idx = blockIdx.x * blockDim.x + threadIdx.x;       // over TOKD/4
    if (idx >= TOKD / 4) return;
    int d4 = (idx & 31) * 4;
    int s  = (idx >> 5) & (Sdim - 1);
    int b  = idx >> 14;
    float wr[4][Kdim];
#pragma unroll
    for (int c = 0; c < 4; ++c)
#pragma unroll
        for (int j = 0; j < Kdim; ++j) wr[c][j] = w[(d4 + c) * Kdim + j];
    float4 acc = { bias[d4], bias[d4 + 1], bias[d4 + 2], bias[d4 + 3] };
#pragma unroll
    for (int j = 0; j < Kdim; ++j) {
        int sp = s + j - Kdim / 2;
        if (sp >= 0 && sp < Sdim) {
            float4 hv = *(const float4*)(h + ((size_t)b * Sdim + sp) * Ddim + d4);
            acc.x += hv.x * wr[0][j];
            acc.y += hv.y * wr[1][j];
            acc.z += hv.z * wr[2][j];
            acc.w += hv.w * wr[3][j];
        }
    }
    v4h h4 = { (_Float16)acc.x, (_Float16)acc.y, (_Float16)acc.z, (_Float16)acc.w };
    *(v4h*)(out + (size_t)idx * 4) = h4;
}

// ---------------------------------------------------------------------------
// Weight packing: logical B[K=128][N=128] -> fragment-major f16:
//   dst[((tileN*4 + k0)*32 + lane)*16 + e]
// so each lane's half-fragment is a contiguous b128 load in the GEMM.
// mode 0: B[k][n] = src[n*128+k]   (pw, f1, f2 : transpose)
// mode 1: B[k][n] = src[(h*128+k)*16+kk], n = h*16+kk   (Wq/Wk/Wv)
// mode 2: B[k][n] = src[k*128+n]   (Wo : copy)
// ---------------------------------------------------------------------------
__global__ void pack_kernel(const float* __restrict__ src, _Float16* __restrict__ dst, int mode) {
    int idx = blockIdx.x * blockDim.x + threadIdx.x;       // 16384
    if (idx >= Ddim * Ddim) return;
    int e    = idx & 15;
    int lane = (idx >> 4) & 31;
    int k0   = (idx >> 9) & 3;
    int tn   = idx >> 11;
    int khalf = lane >> 4, nloc = lane & 15;
    int v = e >> 1, bit = e & 1;
    int k = k0 * 32 + ((v < 4) ? (v * 2) : (16 + (v - 4) * 2)) + khalf * 8 + bit;
    int n = tn * 16 + nloc;
    float val;
    if (mode == 0)      val = src[n * Ddim + k];
    else if (mode == 2) val = src[k * Ddim + n];
    else { int hh = n >> 4; int kk = n & 15; val = src[(hh * Ddim + k) * 16 + kk]; }
    dst[idx] = (_Float16)val;
}

// ---------------------------------------------------------------------------
// GEMM: C[M,128] = A[M,128](f16) x B[128,128](f16 frag-major), 16x64 per wave.
// Epilogue: + bias[col] + (*sbias); relu; + res; write f32 and/or f16
// (f16 optionally in head-transposed [b,h,dk,s] layout for attention V).
// ---------------------------------------------------------------------------
__global__ __launch_bounds__(32)
void gemm128_kernel(const _Float16* __restrict__ A, const _Float16* __restrict__ Bp,
                    const float* __restrict__ bias, const float* __restrict__ sbias,
                    const float* __restrict__ res, float* __restrict__ outf,
                    _Float16* __restrict__ outh, int relu, int outh_trans) {
    int tileM = blockIdx.x;                 // M/16 tiles
    int ng    = blockIdx.y;                 // 0..1 : group of 4 N-tiles
    int lane  = threadIdx.x;
    int nloc  = lane & 15;
    int khalf = lane >> 4;

    const _Float16* Arow = A + (size_t)(tileM * 16 + nloc) * Ddim;
    v8f acc[4] = {};
#pragma unroll
    for (int k0 = 0; k0 < 4; ++k0) {
        H16 a;
        a.h[0] = *(const v8h*)(Arow + k0 * 32 + khalf * 8);
        a.h[1] = *(const v8h*)(Arow + k0 * 32 + 16 + khalf * 8);
#pragma unroll
        for (int t = 0; t < 4; ++t) {
            const _Float16* bp = Bp + ((size_t)((ng * 4 + t) * 4 + k0) * 32 + lane) * 16;
            H16 bf;
            bf.h[0] = *(const v8h*)(bp);
            bf.h[1] = *(const v8h*)(bp + 8);
            acc[t] = __builtin_amdgcn_wmma_f32_16x16x32_f16(false, a.f, false, bf.f,
                                                            (short)0, acc[t], false, false);
        }
    }
    float sb = sbias ? sbias[0] : 0.0f;
    int rowbase = tileM * 16 + (khalf ? 8 : 0);
#pragma unroll
    for (int t = 0; t < 4; ++t) {
        int col = (ng * 4 + t) * 16 + nloc;
        float bv = (bias ? bias[col] : 0.0f) + sb;
#pragma unroll
        for (int r = 0; r < 8; ++r) {
            size_t row = (size_t)(rowbase + r);
            size_t idx = row * Ddim + col;
            float v = acc[t][r] + bv;
            if (relu) v = fmaxf(v, 0.0f);
            if (res)  v += res[idx];
            if (outf) outf[idx] = v;
            if (outh) {
                if (outh_trans) {
                    size_t tidx = ((row >> 9) * Ddim + col) * Sdim + (row & (Sdim - 1));
                    outh[tidx] = (_Float16)v;
                } else {
                    outh[idx] = (_Float16)v;
                }
            }
        }
    }
}

// ---------------------------------------------------------------------------
// Flash attention: one wave per (b, h, 16-query tile). 32 keys per iteration:
// two QK^T WMMAs (DK=16 padded to 32) + one fully-packed K=32 PV WMMA.
// V is consumed in head-transposed layout vT[b][h][dk][s].
// ---------------------------------------------------------------------------
__global__ __launch_bounds__(32)
void fattn_kernel(const _Float16* __restrict__ qh, const _Float16* __restrict__ kh,
                  const _Float16* __restrict__ vT, const int* __restrict__ mask,
                  _Float16* __restrict__ outh) {
    __shared__ _Float16 lp[16][32];
    int bid = blockIdx.x;
    int qt = bid & ((Sdim / 16) - 1);
    int hh = (bid >> 5) & (Hdim - 1);
    int b  = bid >> 8;
    int lane  = threadIdx.x;
    int nloc  = lane & 15;
    int khalf = lane >> 4;
    v8h zero8 = {};

    // Q fragment: channels are contiguous -> single b128; upper K half zero.
    const _Float16* qrow = qh + ((size_t)(b * Sdim) + qt * 16 + nloc) * Ddim + hh * 16;
    H16 aq;
    aq.h[0] = *(const v8h*)(qrow + khalf * 8);
    aq.h[1] = zero8;

    const _Float16* vrow = vT + ((size_t)((b * Hdim + hh) * 16) + nloc) * Sdim;

    float mrow[8], lrow[8];
#pragma unroll
    for (int r = 0; r < 8; ++r) { mrow[r] = -1e30f; lrow[r] = 0.0f; }
    v8f acc = {};

    for (int t0 = 0; t0 < Sdim; t0 += 32) {
        // Two K^T fragments (16 keys each)
        const _Float16* krow0 = kh + ((size_t)(b * Sdim) + t0 + nloc) * Ddim + hh * 16;
        const _Float16* krow1 = krow0 + 16 * Ddim;
        H16 bk0, bk1;
        bk0.h[0] = *(const v8h*)(krow0 + khalf * 8); bk0.h[1] = zero8;
        bk1.h[0] = *(const v8h*)(krow1 + khalf * 8); bk1.h[1] = zero8;
        v8f z = {};
        v8f sc0 = __builtin_amdgcn_wmma_f32_16x16x32_f16(false, aq.f, false, bk0.f,
                                                         (short)0, z, false, false);
        v8f sc1 = __builtin_amdgcn_wmma_f32_16x16x32_f16(false, aq.f, false, bk1.f,
                                                         (short)0, z, false, false);
        float mk0 = (float)mask[b * Sdim + t0 + nloc];
        float mk1 = (float)mask[b * Sdim + t0 + 16 + nloc];
        float p0[8], p1[8];
#pragma unroll
        for (int r = 0; r < 8; ++r) {
            float sv0 = sc0[r] * 0.25f;                      // 1/sqrt(DK)
            float sv1 = sc1[r] * 0.25f;
            sv0 = mk0 * sv0 + (1.0f - mk0) * (-1e30f);
            sv1 = mk1 * sv1 + (1.0f - mk1) * (-1e30f);
            float rm = fmaxf(sv0, sv1);
            rm = fmaxf(rm, __shfl_xor(rm, 1, 32));
            rm = fmaxf(rm, __shfl_xor(rm, 2, 32));
            rm = fmaxf(rm, __shfl_xor(rm, 4, 32));
            rm = fmaxf(rm, __shfl_xor(rm, 8, 32));
            float mnew = fmaxf(mrow[r], rm);
            float corr = __expf(mrow[r] - mnew);
            float pv0 = __expf(sv0 - mnew);
            float pv1 = __expf(sv1 - mnew);
            float ps = pv0 + pv1;
            ps += __shfl_xor(ps, 1, 32);
            ps += __shfl_xor(ps, 2, 32);
            ps += __shfl_xor(ps, 4, 32);
            ps += __shfl_xor(ps, 8, 32);
            lrow[r] = lrow[r] * corr + ps;
            mrow[r] = mnew;
            acc[r] *= corr;
            p0[r] = pv0;
            p1[r] = pv1;
        }
        // Stage P (D-layout -> A-layout) through LDS
        __syncthreads();
#pragma unroll
        for (int r = 0; r < 8; ++r) {
            lp[r + 8 * khalf][nloc]      = (_Float16)p0[r];
            lp[r + 8 * khalf][16 + nloc] = (_Float16)p1[r];
        }
        __syncthreads();
        H16 ap, bv;
        ap.h[0] = *(const v8h*)(&lp[nloc][khalf * 8]);
        ap.h[1] = *(const v8h*)(&lp[nloc][16 + khalf * 8]);
        bv.h[0] = *(const v8h*)(vrow + t0 + khalf * 8);
        bv.h[1] = *(const v8h*)(vrow + t0 + 16 + khalf * 8);
        acc = __builtin_amdgcn_wmma_f32_16x16x32_f16(false, ap.f, false, bv.f,
                                                     (short)0, acc, false, false);
    }

#pragma unroll
    for (int r = 0; r < 8; ++r) {
        int row = qt * 16 + r + 8 * khalf;
        float o = acc[r] / lrow[r];
        outh[((size_t)(b * Sdim) + row) * Ddim + hh * 16 + nloc] = (_Float16)o;
    }
}

// ---------------------------------------------------------------------------
// Host-side launcher
// ---------------------------------------------------------------------------
extern "C" void kernel_launch(void* const* d_in, const int* in_sizes, int n_in,
                              void* d_out, int out_size, void* d_ws, size_t ws_size,
                              hipStream_t stream) {
    (void)in_sizes; (void)n_in; (void)out_size; (void)ws_size;

    const float* x        = (const float*)d_in[0];
    const int*   mask     = (const int*)d_in[1];
    const float* ln_scale = (const float*)d_in[2];
    const float* ln_bias  = (const float*)d_in[3];
    const float* dw_w     = (const float*)d_in[4];
    const float* dw_b     = (const float*)d_in[5];
    const float* pw_w     = (const float*)d_in[6];
    const float* pw_b     = (const float*)d_in[7];
    const float* Wq       = (const float*)d_in[8];
    const float* Wk       = (const float*)d_in[9];
    const float* Wv       = (const float*)d_in[10];
    const float* Wo       = (const float*)d_in[11];
    const float* att_bias = (const float*)d_in[12];
    const float* f1_w     = (const float*)d_in[13];
    const float* f1_b     = (const float*)d_in[14];
    const float* f2_w     = (const float*)d_in[15];
    const float* f2_b     = (const float*)d_in[16];

    // Workspace layout
    char* p = (char*)d_ws;
    float*     cur  = (float*)p;      p += (size_t)TOKD * sizeof(float);
    float*     tmp  = (float*)p;      p += (size_t)TOKD * sizeof(float);
    _Float16*  a16  = (_Float16*)p;   p += (size_t)TOKD * sizeof(_Float16);
    _Float16*  qh   = (_Float16*)p;   p += (size_t)TOKD * sizeof(_Float16);
    _Float16*  kh   = (_Float16*)p;   p += (size_t)TOKD * sizeof(_Float16);
    _Float16*  vh   = (_Float16*)p;   p += (size_t)TOKD * sizeof(_Float16);   // head-transposed V
    _Float16*  atth = (_Float16*)p;   p += (size_t)TOKD * sizeof(_Float16);
    _Float16*  wpw  = (_Float16*)p;   p += (size_t)NCdim * Ddim * Ddim * sizeof(_Float16);
    _Float16*  wq16 = (_Float16*)p;   p += (size_t)Ddim * Ddim * sizeof(_Float16);
    _Float16*  wk16 = (_Float16*)p;   p += (size_t)Ddim * Ddim * sizeof(_Float16);
    _Float16*  wv16 = (_Float16*)p;   p += (size_t)Ddim * Ddim * sizeof(_Float16);
    _Float16*  wo16 = (_Float16*)p;   p += (size_t)Ddim * Ddim * sizeof(_Float16);
    _Float16*  wf1  = (_Float16*)p;   p += (size_t)Ddim * Ddim * sizeof(_Float16);
    _Float16*  wf2  = (_Float16*)p;   p += (size_t)Ddim * Ddim * sizeof(_Float16);

    const int DD = Ddim * Ddim;                       // 16384
    dim3 packGrid((DD + 255) / 256), packBlk(256);
    dim3 vecGrid((TOKD / 4 + 255) / 256), vecBlk(256);
    dim3 gemmGrid(TOKS / 16, 2), gemmBlk(32);         // 512 x 2 (16x64 tiles), 1 wave each

    // ---- pack weights to fragment-major f16 ----
    for (int i = 0; i < NCdim; ++i)
        pack_kernel<<<packGrid, packBlk, 0, stream>>>(pw_w + (size_t)i * DD, wpw + (size_t)i * DD, 0);
    pack_kernel<<<packGrid, packBlk, 0, stream>>>(Wq, wq16, 1);
    pack_kernel<<<packGrid, packBlk, 0, stream>>>(Wk, wk16, 1);
    pack_kernel<<<packGrid, packBlk, 0, stream>>>(Wv, wv16, 1);
    pack_kernel<<<packGrid, packBlk, 0, stream>>>(Wo, wo16, 2);
    pack_kernel<<<packGrid, packBlk, 0, stream>>>(f1_w, wf1, 0);
    pack_kernel<<<packGrid, packBlk, 0, stream>>>(f2_w, wf2, 0);

    // ---- positional signal ----
    pos_add_kernel<<<vecGrid, vecBlk, 0, stream>>>(x, cur);

    // ---- depthwise-separable conv stack ----
    for (int i = 0; i < NCdim; ++i) {
        ln_kernel<<<TOKS, 32, 0, stream>>>(cur, ln_scale + (size_t)i * Ddim,
                                           ln_bias + (size_t)i * Ddim, tmp, nullptr);
        dwconv_kernel<<<vecGrid, vecBlk, 0, stream>>>(tmp, dw_w + (size_t)i * Ddim * Kdim,
                                                      dw_b + (size_t)i * Ddim, a16);
        // relu(pw(x)+b) + residual, in-place on cur
        gemm128_kernel<<<gemmGrid, gemmBlk, 0, stream>>>(a16, wpw + (size_t)i * DD,
                                                         pw_b + (size_t)i * Ddim, nullptr,
                                                         cur, cur, nullptr, 1, 0);
    }

    // ---- multi-head self-attention ----
    ln_kernel<<<TOKS, 32, 0, stream>>>(cur, ln_scale + (size_t)NCdim * Ddim,
                                       ln_bias + (size_t)NCdim * Ddim, nullptr, a16);
    gemm128_kernel<<<gemmGrid, gemmBlk, 0, stream>>>(a16, wq16, nullptr, att_bias,
                                                     nullptr, nullptr, qh, 0, 0);
    gemm128_kernel<<<gemmGrid, gemmBlk, 0, stream>>>(a16, wk16, nullptr, att_bias,
                                                     nullptr, nullptr, kh, 0, 0);
    gemm128_kernel<<<gemmGrid, gemmBlk, 0, stream>>>(a16, wv16, nullptr, att_bias,
                                                     nullptr, nullptr, vh, 0, 1);   // transposed V
    fattn_kernel<<<Bdim * Hdim * (Sdim / 16), 32, 0, stream>>>(qh, kh, vh, mask, atth);
    gemm128_kernel<<<gemmGrid, gemmBlk, 0, stream>>>(atth, wo16, nullptr, att_bias,
                                                     cur, cur, nullptr, 0, 0);

    // ---- feed-forward ----
    ln_kernel<<<TOKS, 32, 0, stream>>>(cur, ln_scale + (size_t)(NCdim + 1) * Ddim,
                                       ln_bias + (size_t)(NCdim + 1) * Ddim, nullptr, a16);
    gemm128_kernel<<<gemmGrid, gemmBlk, 0, stream>>>(a16, wf1, f1_b, nullptr,
                                                     nullptr, nullptr, qh, 1, 0);
    gemm128_kernel<<<gemmGrid, gemmBlk, 0, stream>>>(qh, wf2, f2_b, nullptr,
                                                     cur, (float*)d_out, nullptr, 0, 0);
}
